// MessagePassingLayer_28887950033284
// MI455X (gfx1250) — matrix operand order repaired and checked by reading, not compiled, not run on hardware
//
#include <hip/hip_runtime.h>
#include <hip/hip_bf16.h>

// ---- CDNA5 WMMA types ----
typedef __attribute__((ext_vector_type(16))) __bf16 v16bf;
typedef __attribute__((ext_vector_type(8)))  __bf16 v8bf;
typedef __attribute__((ext_vector_type(8)))  float  v8f;
typedef __attribute__((ext_vector_type(4)))  float  v4f;

#define NODE_DIM 64
#define HIDDEN   64
#define EDGE_DIM 32

// ---------------------------------------------------------------------------
// A-operand builder from an f32 row: 16x32 bf16 tile, row = lane%16.
// Per ISA 7.12.2 (16-bit A 16x32):
//   lane<16 : V0..V3 = K 0..7,  V4..V7 = K 16..23
//   lane>=16: V0..V3 = K 8..15, V4..V7 = K 24..31
// b1 = 0 (lanes 0-15) or 8 (lanes 16-31). NT marks single-use streams.
// ---------------------------------------------------------------------------
template <bool NT>
__device__ __forceinline__ v16bf load_a16_f32(const float* rp, int cb, int b1) {
  const v4f* q0 = (const v4f*)(rp + cb + b1);
  const v4f* q1 = (const v4f*)(rp + cb + 16 + b1);
  v4f f0, f1, f2, f3;
  if (NT) {
    f0 = __builtin_nontemporal_load(q0);
    f1 = __builtin_nontemporal_load(q0 + 1);
    f2 = __builtin_nontemporal_load(q1);
    f3 = __builtin_nontemporal_load(q1 + 1);
  } else {
    f0 = q0[0]; f1 = q0[1]; f2 = q1[0]; f3 = q1[1];
  }
  v16bf a;
#pragma unroll
  for (int i = 0; i < 4; ++i) {
    a[i]      = (__bf16)f0[i];
    a[4 + i]  = (__bf16)f1[i];
    a[8 + i]  = (__bf16)f2[i];
    a[12 + i] = (__bf16)f3[i];
  }
  return a;
}

// A-operand builder from a pre-converted bf16 row: two aligned 16B loads,
// zero VALU conversion work in the hot loop.
__device__ __forceinline__ v16bf load_a16_bf(const __bf16* rp, int cb, int b1) {
  v8bf lo = *(const v8bf*)(rp + cb + b1);
  v8bf hi = *(const v8bf*)(rp + cb + 16 + b1);
  v16bf a;
#pragma unroll
  for (int i = 0; i < 8; ++i) { a[i] = lo[i]; a[8 + i] = hi[i]; }
  return a;
}

__device__ __forceinline__ v8f wmma_bf16(v16bf a, v16bf b, v8f c) {
  return __builtin_amdgcn_wmma_f32_16x16x32_bf16(false, a, false, b,
                                                 (short)0, c, false, false);
}

// ---------------------------------------------------------------------------
// Weight packing: W (fan_in x 64, row-major f32) -> bf16 B-operand tiles.
// Tile (kc, nt) covers K = kc*32..+31, N = nt*16..+15. Per-lane 16 bf16:
//   lane<16 : K = kc*32 + {0..7,16..23}, lane>=16: K = kc*32 + {8..15,24..31}
//   col N = nt*16 + lane%16. Stored [tile][lane][i]: 32 contiguous B per lane.
// ---------------------------------------------------------------------------
__device__ __forceinline__ void pack_one(const float* W, __bf16* out, int p) {
  int tile = p >> 9, r = p & 511;
  int lane = r >> 4, i = r & 15;
  int kc = tile >> 2, nt = tile & 3;
  int k = kc * 32 + ((lane < 16) ? 0 : 8) + ((i < 8) ? i : i + 8);
  int n = nt * 16 + (lane & 15);
  out[p] = (__bf16)W[(size_t)k * 64 + n];
}

__global__ void mp_pack_weights(const float* __restrict__ eW1, const float* __restrict__ eW2,
                                const float* __restrict__ uW1, const float* __restrict__ uW2,
                                __bf16* __restrict__ p1, __bf16* __restrict__ p2,
                                __bf16* __restrict__ p3, __bf16* __restrict__ p4) {
  int idx = blockIdx.x * blockDim.x + threadIdx.x;
  if (idx < 10240)        pack_one(eW1, p1, idx);              // 20 tiles (K=160)
  else if (idx < 14336)   pack_one(eW2, p2, idx - 10240);      //  8 tiles (K=64)
  else if (idx < 22528)   pack_one(uW1, p3, idx - 14336);      // 16 tiles (K=128)
  else if (idx < 26624)   pack_one(uW2, p4, idx - 22528);      //  8 tiles (K=64)
}

__global__ void mp_zero(v4f* __restrict__ p, long long n4) {
  long long i = (long long)blockIdx.x * blockDim.x + threadIdx.x;
  long long stride = (long long)gridDim.x * blockDim.x;
  v4f z = {0.f, 0.f, 0.f, 0.f};
  for (; i < n4; i += stride) p[i] = z;
}

// One-time h f32 -> bf16 copy (L2-hot table gathered ~32x by the edge pass).
__global__ void mp_cvt_h(const float* __restrict__ h, __bf16* __restrict__ hb,
                         long long n8) {
  long long i = (long long)blockIdx.x * blockDim.x + threadIdx.x;
  long long stride = (long long)gridDim.x * blockDim.x;
  const v4f* src = (const v4f*)h;
  for (; i < n8; i += stride) {
    v4f a = src[2 * i], b = src[2 * i + 1];
    v8bf o;
#pragma unroll
    for (int j = 0; j < 4; ++j) { o[j] = (__bf16)a[j]; o[4 + j] = (__bf16)b[j]; }
    ((v8bf*)hb)[i] = o;
  }
}

// ---------------------------------------------------------------------------
// Edge kernel: per wave, 16 edges.
//   inp = [h[src] | h[tgt] | edge_feat]  (16 x 160)
//   msg = relu(inp@eW1 + eb1) @ eW2 + eb2   -> atomic scatter-add into agg[tgt]
// ---------------------------------------------------------------------------
__global__ __launch_bounds__(256) void mp_edge_kernel(
    const __bf16* __restrict__ hb, const int* __restrict__ ei,
    const float* __restrict__ ef,
    const __bf16* __restrict__ B1, const __bf16* __restrict__ B2,
    const float* __restrict__ eb1, const float* __restrict__ eb2,
    float* __restrict__ agg, int E) {
  __shared__ float sbuf[8 * 16 * 64];  // per-wave 16x64 relu slab (32 KB)
  const int lane   = threadIdx.x & 31;
  const int wave   = threadIdx.x >> 5;
  const int col    = lane & 15;        // edge row / N column within tile
  const int rowoff = (lane >> 4) << 3; // 0 or 8: A K-half selector & D row half
  float* my = sbuf + wave * 1024;

  // Hoist weight B-tiles into registers (28 x 8 VGPRs); reused every tile.
  v16bf b1t[20];
#pragma unroll
  for (int t = 0; t < 20; ++t) b1t[t] = *(const v16bf*)(B1 + t * 512 + lane * 16);
  v16bf b2t[8];
#pragma unroll
  for (int t = 0; t < 8; ++t)  b2t[t] = *(const v16bf*)(B2 + t * 512 + lane * 16);
  float bias1[4], bias2[4];
#pragma unroll
  for (int nt = 0; nt < 4; ++nt) { bias1[nt] = eb1[nt * 16 + col]; bias2[nt] = eb2[nt * 16 + col]; }

  const int ntiles  = E >> 4;
  const int wstride = (gridDim.x * blockDim.x) >> 5;
  for (int tile = (blockIdx.x * blockDim.x + threadIdx.x) >> 5; tile < ntiles;
       tile += wstride) {
    const int e    = tile * 16 + col;
    const int srcI = __builtin_nontemporal_load(ei + e);
    const int tgtI = __builtin_nontemporal_load(ei + E + e);

    // prefetch next tile's contiguous edge_feat block (2 KB)
    int nxt = tile + wstride;
    if (nxt < ntiles) __builtin_prefetch(ef + (size_t)nxt * 16 * EDGE_DIM, 0, 1);

    v8f acc[4];
#pragma unroll
    for (int nt = 0; nt < 4; ++nt) {
      float b = bias1[nt];
      acc[nt] = (v8f){b, b, b, b, b, b, b, b};
    }
    // GEMM1: K = 160 in 5 chunks of 32
#pragma unroll
    for (int kc = 0; kc < 5; ++kc) {
      v16bf a;
      if (kc < 4) {
        const __bf16* rp = hb + (size_t)((kc < 2) ? srcI : tgtI) * NODE_DIM;
        a = load_a16_bf(rp, (kc & 1) ? 32 : 0, rowoff);
      } else {
        a = load_a16_f32<true>(ef + (size_t)e * EDGE_DIM, 0, rowoff);
      }
#pragma unroll
      for (int nt = 0; nt < 4; ++nt)
        acc[nt] = wmma_bf16(a, b1t[kc * 4 + nt], acc[nt]);
    }
    // ReLU, re-layout D->A via per-wave LDS slab
#pragma unroll
    for (int nt = 0; nt < 4; ++nt)
#pragma unroll
      for (int v = 0; v < 8; ++v)
        my[(v + rowoff) * 64 + nt * 16 + col] = fmaxf(acc[nt][v], 0.f);

    // GEMM2: K = 64 in 2 chunks
    v8f acc2[4];
#pragma unroll
    for (int nt = 0; nt < 4; ++nt) {
      float b = bias2[nt];
      acc2[nt] = (v8f){b, b, b, b, b, b, b, b};
    }
#pragma unroll
    for (int kc = 0; kc < 2; ++kc) {
      v16bf a = load_a16_f32<false>(my + col * 64, kc * 32, rowoff);
#pragma unroll
      for (int nt = 0; nt < 4; ++nt)
        acc2[nt] = wmma_bf16(a, b2t[kc * 4 + nt], acc2[nt]);
    }
    // Scatter-add messages into agg[tgt] (agg is L2-resident, 25.6 MB)
#pragma unroll
    for (int v = 0; v < 8; ++v) {
      const int m = v + rowoff;                 // edge row this VGPR holds
      const int t = __shfl(tgtI, m, 32);        // tgt of that edge
#pragma unroll
      for (int nt = 0; nt < 4; ++nt)
        (void)__hip_atomic_fetch_add(agg + (size_t)t * HIDDEN + nt * 16 + col,
                                     acc2[nt][v], __ATOMIC_RELAXED,
                                     __HIP_MEMORY_SCOPE_AGENT);
    }
  }
}

// ---------------------------------------------------------------------------
// Update kernel: per wave, 16 nodes.
//   out = relu([h | agg] @ uW1 + ub1) @ uW2 + ub2
// ---------------------------------------------------------------------------
__global__ __launch_bounds__(256) void mp_update_kernel(
    const __bf16* __restrict__ hb, const float* __restrict__ agg,
    const __bf16* __restrict__ B1, const __bf16* __restrict__ B2,
    const float* __restrict__ ub1, const float* __restrict__ ub2,
    float* __restrict__ out, int N) {
  __shared__ float sbuf[8 * 16 * 64];
  const int lane   = threadIdx.x & 31;
  const int wave   = threadIdx.x >> 5;
  const int col    = lane & 15;
  const int rowoff = (lane >> 4) << 3;
  float* my = sbuf + wave * 1024;

  v16bf b1t[16];
#pragma unroll
  for (int t = 0; t < 16; ++t) b1t[t] = *(const v16bf*)(B1 + t * 512 + lane * 16);
  v16bf b2t[8];
#pragma unroll
  for (int t = 0; t < 8; ++t)  b2t[t] = *(const v16bf*)(B2 + t * 512 + lane * 16);
  float bias1[4], bias2[4];
#pragma unroll
  for (int nt = 0; nt < 4; ++nt) { bias1[nt] = ub1[nt * 16 + col]; bias2[nt] = ub2[nt * 16 + col]; }

  const int ntiles  = N >> 4;
  const int wstride = (gridDim.x * blockDim.x) >> 5;
  for (int tile = (blockIdx.x * blockDim.x + threadIdx.x) >> 5; tile < ntiles;
       tile += wstride) {
    const int node = tile * 16 + col;
    const __bf16* hr = hb  + (size_t)node * NODE_DIM;
    const float*  ar = agg + (size_t)node * HIDDEN;

    v8f acc[4];
#pragma unroll
    for (int nt = 0; nt < 4; ++nt) {
      float b = bias1[nt];
      acc[nt] = (v8f){b, b, b, b, b, b, b, b};
    }
    // GEMM1: K = 128 = [h(0:32) | h(32:64) | agg(0:32) | agg(32:64)]
#pragma unroll
    for (int kc = 0; kc < 4; ++kc) {
      v16bf a = (kc < 2) ? load_a16_bf(hr, (kc & 1) ? 32 : 0, rowoff)
                         : load_a16_f32<false>(ar, (kc & 1) ? 32 : 0, rowoff);
#pragma unroll
      for (int nt = 0; nt < 4; ++nt)
        acc[nt] = wmma_bf16(a, b1t[kc * 4 + nt], acc[nt]);
    }
#pragma unroll
    for (int nt = 0; nt < 4; ++nt)
#pragma unroll
      for (int v = 0; v < 8; ++v)
        my[(v + rowoff) * 64 + nt * 16 + col] = fmaxf(acc[nt][v], 0.f);

    v8f acc2[4];
#pragma unroll
    for (int nt = 0; nt < 4; ++nt) {
      float b = bias2[nt];
      acc2[nt] = (v8f){b, b, b, b, b, b, b, b};
    }
#pragma unroll
    for (int kc = 0; kc < 2; ++kc) {
      v16bf a = load_a16_f32<false>(my + col * 64, kc * 32, rowoff);
#pragma unroll
      for (int nt = 0; nt < 4; ++nt)
        acc2[nt] = wmma_bf16(a, b2t[kc * 4 + nt], acc2[nt]);
    }
    // Streaming NT stores: 'out' is written once, never re-read.
#pragma unroll
    for (int v = 0; v < 8; ++v) {
      const size_t row = (size_t)(tile * 16 + v + rowoff) * HIDDEN;
#pragma unroll
      for (int nt = 0; nt < 4; ++nt)
        __builtin_nontemporal_store(acc2[nt][v], out + row + nt * 16 + col);
    }
  }
}

// ---------------------------------------------------------------------------
extern "C" void kernel_launch(void* const* d_in, const int* in_sizes, int n_in,
                              void* d_out, int out_size, void* d_ws, size_t ws_size,
                              hipStream_t stream) {
  const float* h   = (const float*)d_in[0];
  const int*   ei  = (const int*)  d_in[1];
  const float* ef  = (const float*)d_in[2];
  const float* eW1 = (const float*)d_in[3];
  const float* eb1 = (const float*)d_in[4];
  const float* eW2 = (const float*)d_in[5];
  const float* eb2 = (const float*)d_in[6];
  const float* uW1 = (const float*)d_in[7];
  const float* ub1 = (const float*)d_in[8];
  const float* uW2 = (const float*)d_in[9];
  const float* ub2 = (const float*)d_in[10];

  const int N = in_sizes[0] / NODE_DIM;
  const int E = in_sizes[1] / 2;

  char* ws = (char*)d_ws;
  float* agg = (float*)ws;
  size_t off = ((size_t)N * HIDDEN * sizeof(float) + 255) & ~(size_t)255;
  __bf16* hb = (__bf16*)(ws + off);
  off += ((size_t)N * NODE_DIM * sizeof(__bf16) + 255) & ~(size_t)255;
  __bf16* p1 = (__bf16*)(ws + off); off += 20 * 512 * sizeof(__bf16);
  __bf16* p2 = (__bf16*)(ws + off); off += 8  * 512 * sizeof(__bf16);
  __bf16* p3 = (__bf16*)(ws + off); off += 16 * 512 * sizeof(__bf16);
  __bf16* p4 = (__bf16*)(ws + off); off += 8  * 512 * sizeof(__bf16);

  mp_zero<<<2048, 256, 0, stream>>>((v4f*)agg, (long long)N * (HIDDEN / 4));
  mp_cvt_h<<<2048, 256, 0, stream>>>(h, hb, (long long)N * (NODE_DIM / 8));
  mp_pack_weights<<<104, 256, 0, stream>>>(eW1, eW2, uW1, uW2, p1, p2, p3, p4);
  mp_edge_kernel<<<2048, 256, 0, stream>>>(hb, ei, ef, p1, p2, eb1, eb2, agg, E);
  mp_update_kernel<<<512, 256, 0, stream>>>(hb, agg, p3, p4, ub1, ub2,
                                            (float*)d_out, N);
}
